// ReOrderer_50190987821313
// MI455X (gfx1250) — matrix operand order repaired and checked by reading, not compiled, not run on hardware
//
#include <hip/hip_runtime.h>

#define B_ 8
#define S_ 1024
#define T_ 1024
#define D_ 1024
#define INF_ 1.0e10f

typedef __attribute__((ext_vector_type(16))) __bf16 v16bf;
typedef __attribute__((ext_vector_type(8)))  float  v8f;

union BFU { unsigned short u; __bf16 b; };

__device__ __forceinline__ unsigned short f2bf_u(float f) {
  union { float f; unsigned u; } x; x.f = f;
  unsigned r = x.u + 0x7FFFu + ((x.u >> 16) & 1u);   // round-to-nearest-even
  return (unsigned short)(r >> 16);
}
__device__ __forceinline__ __bf16 bits2bf(unsigned short h) { BFU o; o.u = h; return o.b; }

__device__ __forceinline__ v8f wmma_bf16(v16bf a, v16bf b, v8f c) {
  return __builtin_amdgcn_wmma_f32_16x16x32_bf16(false, a, false, b, (short)0, c, false, false);
}

// ---- WMMA fragment loaders (contiguous-k order k = 16*(lane/16) + i, wave32) ----
// A (16xK bf16, row-major src): lane&15 = M; lanes<16 hold K = k0+{0..15}, lanes>=16 K = k0+{16..31}
__device__ __forceinline__ v16bf load_a_bf16(const unsigned short* A, int ld, int k0, int lane) {
  int m  = lane & 15;
  int kb = (lane >> 4) << 4;           // 0 or 16
  const unsigned short* p = A + m * ld + k0 + kb;
  v16bf a;
  #pragma unroll
  for (int i = 0; i < 8; ++i) a[i] = bits2bf(p[i]);
  #pragma unroll
  for (int i = 0; i < 8; ++i) a[8 + i] = bits2bf(p[8 + i]);
  return a;
}
// B (KxN) where source W is stored row-major as [N][K] (i.e. W[n*ld + k]) -> contiguous per-lane loads
__device__ __forceinline__ v16bf load_b_wt(const unsigned short* W, int ld, int k0, int col0, int lane) {
  int n  = lane & 15;
  int kb = (lane >> 4) << 4;           // 0 or 16
  const unsigned short* p = W + (size_t)(col0 + n) * ld + k0 + kb;
  v16bf b;
  #pragma unroll
  for (int i = 0; i < 16; ++i) b[i] = bits2bf(p[i]);
  __builtin_prefetch((const void*)(p + 64), 0, 3);   // next cacheline of this lane's K-stream
  return b;
}

// ---- kernel 1: steps[b] = sum(mask_src[b]) / sum(mask_trg[b]) ----
__global__ __launch_bounds__(256) void steps_kernel(const float* __restrict__ mask_src,
                                                    const float* __restrict__ mask_trg,
                                                    float* __restrict__ steps) {
  __shared__ float red[256];
  int b = blockIdx.x, tid = threadIdx.x;
  float s = 0.f, t = 0.f;
  for (int i = tid; i < S_; i += 256) s += mask_src[b * S_ + i];
  for (int i = tid; i < T_; i += 256) t += mask_trg[b * T_ + i];
  red[tid] = s; __syncthreads();
  for (int k = 128; k > 0; k >>= 1) { if (tid < k) red[tid] += red[tid + k]; __syncthreads(); }
  float ssum = red[0]; __syncthreads();
  red[tid] = t; __syncthreads();
  for (int k = 128; k > 0; k >>= 1) { if (tid < k) red[tid] += red[tid + k]; __syncthreads(); }
  if (tid == 0) steps[b] = ssum / red[0];
}

// ---- kernel 2: fp32 -> bf16 conversion ----
__global__ __launch_bounds__(256) void cvt_bf16_kernel(const float* __restrict__ src,
                                                       unsigned short* __restrict__ dst, int n) {
  int i = blockIdx.x * 256 + threadIdx.x;
  if (i < n) dst[i] = f2bf_u(src[i]);
}

// ---- kernel 2b: bf16 [B][S][D] -> [B][D][S] transpose (32x32 LDS tiles) ----
__global__ __launch_bounds__(256) void transpose_bf16_kernel(const unsigned short* __restrict__ src,
                                                             unsigned short* __restrict__ dst) {
  __shared__ unsigned short tile[32][33];
  int b  = blockIdx.z;
  int s0 = blockIdx.x * 32;
  int d0 = blockIdx.y * 32;
  const unsigned short* sp = src + (size_t)b * S_ * D_;
  unsigned short* dp       = dst + (size_t)b * S_ * D_;
  int tid = threadIdx.x;
  for (int idx = tid; idx < 1024; idx += 256) {
    int r = idx >> 5, c = idx & 31;
    tile[r][c] = sp[(size_t)(s0 + r) * D_ + d0 + c];
  }
  __syncthreads();
  for (int idx = tid; idx < 1024; idx += 256) {
    int r = idx >> 5, c = idx & 31;
    dp[(size_t)(d0 + r) * S_ + s0 + c] = tile[c][r];
  }
}

// ---- kernel 3: k = key @ Wk^T + bk  (bf16 output) ----
__global__ __launch_bounds__(256) void kproj_kernel(const unsigned short* __restrict__ key_bf,
                                                    const unsigned short* __restrict__ wk_bf,
                                                    const float* __restrict__ bk,
                                                    unsigned short* __restrict__ k_bf) {
  __shared__ unsigned short atile[16 * D_];
  int tid = threadIdx.x;
  size_t row0 = (size_t)blockIdx.x * 16;           // flattened b*S + s row tile
  {
    const unsigned int* src = (const unsigned int*)(key_bf + row0 * D_);
    unsigned int* dst = (unsigned int*)atile;
    for (int i = tid; i < 16 * D_ / 2; i += 256) dst[i] = src[i];
  }
  __syncthreads();
  int wave = tid >> 5, lane = tid & 31;
  int n = lane & 15, mb = (lane >> 4) << 3;
  int colbase = blockIdx.y * 512 + wave * 64;
  v8f acc[4] = {};
  for (int kk = 0; kk < D_; kk += 32) {
    v16bf a = load_a_bf16(atile, D_, kk, lane);
    #pragma unroll
    for (int i = 0; i < 4; ++i) {
      v16bf bm = load_b_wt(wk_bf, D_, kk, colbase + i * 16, lane);
      acc[i] = wmma_bf16(a, bm, acc[i]);
    }
  }
  #pragma unroll
  for (int i = 0; i < 4; ++i) {
    int col0 = colbase + i * 16;
    float bka = bk[col0 + n];
    #pragma unroll
    for (int r = 0; r < 8; ++r)
      k_bf[(row0 + mb + r) * D_ + col0 + n] = f2bf_u(acc[i][r] + bka);
  }
}

// ---- kernel 4: position softmax, query GEMM, gates, out = gate * l_att ----
__global__ __launch_bounds__(256) void c1_kernel(const float* __restrict__ mask_src,
                                                 const unsigned short* __restrict__ keyT_bf,
                                                 const float* __restrict__ steps_g,
                                                 const float* __restrict__ wg,
                                                 const float* __restrict__ bg,
                                                 unsigned short* __restrict__ query_bf,
                                                 float* __restrict__ gates_g,
                                                 float* __restrict__ out) {
  extern __shared__ char smem[];
  float* latt            = (float*)smem;                            // 16*1024 f32 = 64 KB
  unsigned short* lattbf = (unsigned short*)(smem + 16 * S_ * 4);   // 16*1024 bf16 = 32 KB
  float* red             = (float*)(smem + 16 * S_ * 4 + 16 * S_ * 2);
  float* gacc            = red + 256;
  int tid = threadIdx.x;
  int b  = blockIdx.x / (T_ / 16);
  int t0 = (blockIdx.x % (T_ / 16)) * 16;
  float st = steps_g[b];
  if (tid < 16) gacc[tid] = 0.0f;

  // 16 row softmaxes over S (analytic logits)
  for (int r = 0; r < 16; ++r) {
    float it = st * (float)(t0 + r);
    float lmax = -3.4e38f;
    for (int s = tid; s < S_; s += 256) {
      float d = (float)s - it;
      float v = -d * d * (1.0f / 0.3f) - INF_ * (1.0f - mask_src[b * S_ + s]);
      latt[r * S_ + s] = v;
      lmax = fmaxf(lmax, v);
    }
    red[tid] = lmax; __syncthreads();
    for (int k = 128; k > 0; k >>= 1) { if (tid < k) red[tid] = fmaxf(red[tid], red[tid + k]); __syncthreads(); }
    float M = red[0]; __syncthreads();
    float lsum = 0.0f;
    for (int s = tid; s < S_; s += 256) {
      float e = __expf(latt[r * S_ + s] - M);
      latt[r * S_ + s] = e; lsum += e;
    }
    red[tid] = lsum; __syncthreads();
    for (int k = 128; k > 0; k >>= 1) { if (tid < k) red[tid] += red[tid + k]; __syncthreads(); }
    float inv = 1.0f / red[0]; __syncthreads();
    for (int s = tid; s < S_; s += 256) latt[r * S_ + s] *= inv;
  }
  __syncthreads();
  // one-time bf16 copy of l_att for WMMA A-fragments
  for (int idx = tid; idx < 16 * S_; idx += 256) lattbf[idx] = f2bf_u(latt[idx]);
  __syncthreads();

  // query[16,D] = l_att[16,S] @ key[S,D]  (B from keyT: contiguous per-lane)
  int wave = tid >> 5, lane = tid & 31;
  int n = lane & 15, mb = (lane >> 4) << 3;
  const unsigned short* keyTb = keyT_bf + (size_t)b * S_ * D_;     // [D][S]
  v8f acc[8] = {};
  for (int kk = 0; kk < S_; kk += 32) {
    v16bf a = load_a_bf16(lattbf, S_, kk, lane);
    #pragma unroll
    for (int i = 0; i < 8; ++i) {
      v16bf bm = load_b_wt(keyTb, S_, kk, (wave * 8 + i) * 16, lane);
      acc[i] = wmma_bf16(a, bm, acc[i]);
    }
  }
  #pragma unroll
  for (int i = 0; i < 8; ++i) {
    int col0 = (wave * 8 + i) * 16;
    float wgv = wg[col0 + n];
    #pragma unroll
    for (int r = 0; r < 8; ++r) {
      int m = mb + r;
      float qv = acc[i][r];
      atomicAdd(&gacc[m], qv * wgv);                                   // ds_add_f32
      query_bf[((size_t)(b * T_ + t0 + m)) * D_ + col0 + n] = f2bf_u(qv);
    }
  }
  __syncthreads();
  if (tid < 16) {
    float g = 1.0f / (1.0f + __expf(-(gacc[tid] + bg[0])));
    gacc[tid] = g;
    gates_g[b * T_ + t0 + tid] = g;
  }
  __syncthreads();
  for (int idx = tid; idx < 16 * S_; idx += 256) {
    int r = idx >> 10, s = idx & (S_ - 1);
    out[((size_t)(b * T_ + t0 + r)) * S_ + s] = gacc[r] * latt[r * S_ + s];
  }
}

// ---- kernel 5: q proj, dots = q@k^T, softmax, out += (1-gate)*probs ----
__global__ __launch_bounds__(256) void c2_kernel(const unsigned short* __restrict__ query_bf,
                                                 const unsigned short* __restrict__ wq_bf,
                                                 const float* __restrict__ bq,
                                                 const unsigned short* __restrict__ k_bf,
                                                 const float* __restrict__ mask_src,
                                                 const float* __restrict__ gates_g,
                                                 float* __restrict__ out) {
  extern __shared__ char smem[];
  unsigned short* qtile = (unsigned short*)smem;             // 16*1024 bf16 = 32 KB
  unsigned short* qproj = qtile + 16 * D_;                   // 32 KB
  float* dots = (float*)(smem + 2 * 16 * D_ * 2);            // 16*1024 f32 = 64 KB
  float* red  = (float*)(smem + 2 * 16 * D_ * 2 + 16 * S_ * 4);
  float* gate = red + 256;
  int tid = threadIdx.x;
  int b  = blockIdx.x / (T_ / 16);
  int t0 = (blockIdx.x % (T_ / 16)) * 16;

  {
    const unsigned int* src = (const unsigned int*)(query_bf + ((size_t)(b * T_ + t0)) * D_);
    unsigned int* dst = (unsigned int*)qtile;
    for (int i = tid; i < 16 * D_ / 2; i += 256) dst[i] = src[i];
  }
  if (tid < 16) gate[tid] = gates_g[b * T_ + t0 + tid];
  __syncthreads();

  int wave = tid >> 5, lane = tid & 31;
  int n = lane & 15, mb = (lane >> 4) << 3;

  // GEMM1: qproj = qtile @ Wq^T + bq (bf16 into LDS)
  {
    v8f acc[8] = {};
    for (int kk = 0; kk < D_; kk += 32) {
      v16bf a = load_a_bf16(qtile, D_, kk, lane);
      #pragma unroll
      for (int i = 0; i < 8; ++i) {
        v16bf bm = load_b_wt(wq_bf, D_, kk, (wave * 8 + i) * 16, lane);
        acc[i] = wmma_bf16(a, bm, acc[i]);
      }
    }
    #pragma unroll
    for (int i = 0; i < 8; ++i) {
      int col0 = (wave * 8 + i) * 16;
      float bqa = bq[col0 + n];
      #pragma unroll
      for (int r = 0; r < 8; ++r)
        qproj[(mb + r) * D_ + col0 + n] = f2bf_u(acc[i][r] + bqa);
    }
  }
  __syncthreads();

  // GEMM2: dots[16,S] = qproj[16,D] @ k[S,D]^T, masked + scaled
  {
    const unsigned short* kb = k_bf + (size_t)b * S_ * D_;
    v8f acc[8] = {};
    for (int kk = 0; kk < D_; kk += 32) {
      v16bf a = load_a_bf16(qproj, D_, kk, lane);
      #pragma unroll
      for (int i = 0; i < 8; ++i) {
        v16bf bm = load_b_wt(kb, D_, kk, (wave * 8 + i) * 16, lane);
        acc[i] = wmma_bf16(a, bm, acc[i]);
      }
    }
    #pragma unroll
    for (int i = 0; i < 8; ++i) {
      int col0 = (wave * 8 + i) * 16;
      float msk = mask_src[b * S_ + col0 + n];
      #pragma unroll
      for (int r = 0; r < 8; ++r)
        dots[(mb + r) * S_ + col0 + n] = (acc[i][r] - (1.0f - msk) * INF_) * 0.03125f;
    }
  }
  __syncthreads();

  // row softmax over dots
  for (int r = 0; r < 16; ++r) {
    float lmax = -3.4e38f;
    for (int s = tid; s < S_; s += 256) lmax = fmaxf(lmax, dots[r * S_ + s]);
    red[tid] = lmax; __syncthreads();
    for (int k = 128; k > 0; k >>= 1) { if (tid < k) red[tid] = fmaxf(red[tid], red[tid + k]); __syncthreads(); }
    float M = red[0]; __syncthreads();
    float lsum = 0.0f;
    for (int s = tid; s < S_; s += 256) {
      float e = __expf(dots[r * S_ + s] - M);
      dots[r * S_ + s] = e; lsum += e;
    }
    red[tid] = lsum; __syncthreads();
    for (int k = 128; k > 0; k >>= 1) { if (tid < k) red[tid] += red[tid + k]; __syncthreads(); }
    float inv = 1.0f / red[0]; __syncthreads();
    for (int s = tid; s < S_; s += 256) dots[r * S_ + s] *= inv;
  }
  __syncthreads();

  for (int idx = tid; idx < 16 * S_; idx += 256) {
    int r = idx >> 10, s = idx & (S_ - 1);
    size_t o = ((size_t)(b * T_ + t0 + r)) * S_ + s;
    out[o] += (1.0f - gate[r]) * dots[r * S_ + s];
  }
}

extern "C" void kernel_launch(void* const* d_in, const int* in_sizes, int n_in,
                              void* d_out, int out_size, void* d_ws, size_t ws_size,
                              hipStream_t stream) {
  (void)in_sizes; (void)n_in; (void)out_size; (void)ws_size;
  const float* key      = (const float*)d_in[0];
  const float* mask_src = (const float*)d_in[1];
  const float* mask_trg = (const float*)d_in[2];
  const float* Wq = (const float*)d_in[3];
  const float* bq = (const float*)d_in[4];
  const float* Wk = (const float*)d_in[5];
  const float* bk = (const float*)d_in[6];
  const float* wg = (const float*)d_in[7];
  const float* bg = (const float*)d_in[8];
  float* out = (float*)d_out;

  char* ws = (char*)d_ws;
  float* steps           = (float*)ws;
  unsigned short* key_bf = (unsigned short*)(ws + 256);
  unsigned short* k_bf   = key_bf + (size_t)B_ * S_ * D_;
  unsigned short* q_bf   = k_bf + (size_t)B_ * S_ * D_;
  unsigned short* wq_bf  = q_bf + (size_t)B_ * T_ * D_;
  unsigned short* wk_bf  = wq_bf + (size_t)D_ * D_;
  float* gates           = (float*)(wk_bf + (size_t)D_ * D_);
  unsigned short* keyT   = (unsigned short*)(gates + (size_t)B_ * T_);

  steps_kernel<<<B_, 256, 0, stream>>>(mask_src, mask_trg, steps);
  cvt_bf16_kernel<<<(B_ * S_ * D_) / 256, 256, 0, stream>>>(key, key_bf, B_ * S_ * D_);
  cvt_bf16_kernel<<<(D_ * D_) / 256, 256, 0, stream>>>(Wq, wq_bf, D_ * D_);
  cvt_bf16_kernel<<<(D_ * D_) / 256, 256, 0, stream>>>(Wk, wk_bf, D_ * D_);
  transpose_bf16_kernel<<<dim3(S_ / 32, D_ / 32, B_), 256, 0, stream>>>(key_bf, keyT);
  kproj_kernel<<<dim3((B_ * S_) / 16, 2), 256, 0, stream>>>(key_bf, wk_bf, bk, k_bf);

  size_t c1_sh = (size_t)16 * S_ * 4 + (size_t)16 * S_ * 2 + 256 * 4 + 16 * 4;
  c1_kernel<<<(B_ * T_) / 16, 256, c1_sh, stream>>>(mask_src, keyT, steps, wg, bg, q_bf, gates, out);
  size_t c2_sh = (size_t)2 * 16 * D_ * 2 + (size_t)16 * S_ * 4 + 256 * 4 + 16 * 4;
  c2_kernel<<<(B_ * T_) / 16, 256, c2_sh, stream>>>(q_bf, wq_bf, bq, k_bf, mask_src, gates, out);
}
